// SimpleCausalAttention_52441550684750
// MI455X (gfx1250) — matrix operand order repaired and checked by reading, not compiled, not run on hardware
//
#include <hip/hip_runtime.h>
#include <hip/hip_bf16.h>
#include <stdint.h>

typedef __bf16 bf16;
typedef __attribute__((ext_vector_type(16))) __bf16 v16bf;
typedef __attribute__((ext_vector_type(8)))  __bf16 v8bf;
typedef __attribute__((ext_vector_type(8)))  float  v8f;

// ---------------------------------------------------------------------------
// WMMA fragment loaders (CDNA5 16-bit layouts, wave32)
// A 16x32 (MxK): lane r=lane&15 is row, half=lane>>4; elems K = 8h+c and 16+8h+c
// B 32x16 (KxN): lane n=lane&15 is col, half=lane>>4; elems K = 8h+c and 16+8h+c
//   -> B operand is supplied as B^T stored row-major [N][K]
// C/D 16x16:     row = r + 8*half, col = lane&15 (VGPR r)
// ---------------------------------------------------------------------------
__device__ __forceinline__ v16bf load_a_frag(const bf16* base, int ld, int row0,
                                             int k0, int lane) {
  const int r = lane & 15, h = lane >> 4;
  const bf16* p = base + (size_t)(row0 + r) * ld + k0 + h * 8;
  v8bf lo = *(const v8bf*)p;
  v8bf hi = *(const v8bf*)(p + 16);
  v16bf f;
#pragma unroll
  for (int c = 0; c < 8; ++c) { f[c] = lo[c]; f[c + 8] = hi[c]; }
  return f;
}

__device__ __forceinline__ v16bf load_b_frag(const bf16* baseT, int ld, int col0,
                                             int k0, int lane) {
  const int n = lane & 15, h = lane >> 4;
  const bf16* p = baseT + (size_t)(col0 + n) * ld + k0 + h * 8;
  v8bf lo = *(const v8bf*)p;
  v8bf hi = *(const v8bf*)(p + 16);
  v16bf f;
#pragma unroll
  for (int c = 0; c < 8; ++c) { f[c] = lo[c]; f[c + 8] = hi[c]; }
  return f;
}

__device__ __forceinline__ v8f wmma_bf16(v16bf a, v16bf b, v8f c) {
  return __builtin_amdgcn_wmma_f32_16x16x32_bf16(false, a, false, b, (short)0, c,
                                                 false, false);
}

// ---------------------------------------------------------------------------
// CDNA5 async global->LDS copy (ASYNCcnt-tracked), GVS addressing:
//   mem = SADDR(64) + VADDR(32) + IOFFSET;  LDS dest = VGPR lds byte address
// ---------------------------------------------------------------------------
__device__ __forceinline__ uint32_t lds_off(const void* p) {
  // generic pointer to LDS: low 32 bits carry the workgroup-relative LDS address
  return (uint32_t)(uintptr_t)p;
}

__device__ __forceinline__ void async_copy_b128(uint32_t lds_byte_addr,
                                                const void* sbase,
                                                uint32_t voff_bytes) {
  asm volatile("global_load_async_to_lds_b128 %0, %1, %2"
               :
               : "v"(lds_byte_addr), "v"(voff_bytes), "s"(sbase)
               : "memory");
}

__device__ __forceinline__ void wait_async0() {
  asm volatile("s_wait_asynccnt 0" ::: "memory");
}

// ---------------------------------------------------------------------------
// Elementwise conversion / transpose kernels
// ---------------------------------------------------------------------------
__global__ __launch_bounds__(256) void cast_f32_bf16(const float* __restrict__ in,
                                                     bf16* __restrict__ out,
                                                     long long n) {
  long long i = (long long)blockIdx.x * blockDim.x + threadIdx.x;
  if (i < n) out[i] = (bf16)in[i];
}

// in [rows][cols] fp32 -> out [cols][rows] bf16
__global__ __launch_bounds__(256) void transpose_f32_bf16(const float* __restrict__ in,
                                                          bf16* __restrict__ out,
                                                          int rows, int cols) {
  long long i = (long long)blockIdx.x * blockDim.x + threadIdx.x;
  if (i >= (long long)rows * cols) return;
  int r = (int)(i / cols), c = (int)(i % cols);
  out[(size_t)c * rows + r] = (bf16)in[i];
}

// qkv [token][3*1024] -> q*0.125 [bh][n][64], k [bh][n][64], v^T [bh][64][n]
__global__ __launch_bounds__(256) void reshape_qkv(const bf16* __restrict__ qkv,
                                                   bf16* __restrict__ qh,
                                                   bf16* __restrict__ kh,
                                                   bf16* __restrict__ vt) {
  long long i = (long long)blockIdx.x * blockDim.x + threadIdx.x;  // token*1024+c
  if (i >= (long long)8192 * 1024) return;
  int token = (int)(i >> 10), c = (int)(i & 1023);
  int h = c >> 6, d = c & 63;
  int b = token >> 11, n = token & 2047;
  size_t src = (size_t)token * 3072;
  size_t bh = (size_t)b * 16 + h;
  qh[(bh * 2048 + n) * 64 + d] = (bf16)((float)qkv[src + c] * 0.125f);  // 1/sqrt(64)
  kh[(bh * 2048 + n) * 64 + d] = qkv[src + 1024 + c];
  vt[(bh * 64 + d) * 2048 + n] = qkv[src + 2048 + c];
}

// ---------------------------------------------------------------------------
// Tiled GEMM: C[M,N] = A_bf16[M,K] * B (given as B^T bf16 [N,K])
// 256 threads = 8 waves; wave tile 32x32 (2x2 WMMA); block tile 128x64.
// Double-buffered LDS staging filled by GLOBAL_LOAD_ASYNC_TO_LDS_B128:
// loads for K-step t+1 fly while WMMAs consume step t.
// ---------------------------------------------------------------------------
template <bool OUT_BF16>
__global__ __launch_bounds__(256) void tile_gemm(const bf16* __restrict__ A,
                                                 const bf16* __restrict__ Bt,
                                                 void* __restrict__ Cout,
                                                 int M, int N, int K) {
  const int tid = threadIdx.x;
  const int lane = tid & 31;
  const int wave = tid >> 5;
  const int m0b = blockIdx.y * 128;
  const int n0b = blockIdx.x * 64;
  const int m0 = (wave >> 1) * 32;  // within block tile
  const int n0 = (wave & 1) * 32;

  __shared__ __align__(16) bf16 tA[2][128][32];  // 2 x 8 KB
  __shared__ __align__(16) bf16 tB[2][64][32];   // 2 x 4 KB

  // per-thread 16B copy chunks: A has 512 chunks (2/thread), B has 256 (1/thread)
  const int rA0 = tid >> 2,          cA0 = (tid & 3) * 8;
  const int rA1 = (tid + 256) >> 2,  cA1 = (tid & 3) * 8;
  const int rB  = tid >> 2,          cB  = (tid & 3) * 8;
  const uint32_t gA0 = (uint32_t)(((size_t)(m0b + rA0) * K + cA0) * sizeof(bf16));
  const uint32_t gA1 = (uint32_t)(((size_t)(m0b + rA1) * K + cA1) * sizeof(bf16));
  const uint32_t gB  = (uint32_t)(((size_t)(n0b + rB) * K + cB) * sizeof(bf16));
  const uint32_t lA0 = lds_off(&tA[0][rA0][cA0]);
  const uint32_t lA1 = lds_off(&tA[0][rA1][cA1]);
  const uint32_t lB  = lds_off(&tB[0][rB][cB]);
  const uint32_t bufStrideA = (uint32_t)(128 * 32 * sizeof(bf16));
  const uint32_t bufStrideB = (uint32_t)(64 * 32 * sizeof(bf16));

  auto issue_stage = [&](int buf, int k0) {
    const uint32_t kb = (uint32_t)k0 * sizeof(bf16);
    async_copy_b128(lA0 + buf * bufStrideA, A, gA0 + kb);
    async_copy_b128(lA1 + buf * bufStrideA, A, gA1 + kb);
    async_copy_b128(lB + buf * bufStrideB, Bt, gB + kb);
  };

  v8f acc[2][2] = {};

  issue_stage(0, 0);
  wait_async0();
  __syncthreads();

  for (int k0 = 0; k0 < K; k0 += 32) {
    const int buf = (k0 >> 5) & 1;
    if (k0 + 32 < K) issue_stage(buf ^ 1, k0 + 32);  // overlap next stage

    v16bf a[2], b[2];
    a[0] = load_a_frag(&tA[buf][0][0], 32, m0, 0, lane);
    a[1] = load_a_frag(&tA[buf][0][0], 32, m0 + 16, 0, lane);
    b[0] = load_b_frag(&tB[buf][0][0], 32, n0, 0, lane);
    b[1] = load_b_frag(&tB[buf][0][0], 32, n0 + 16, 0, lane);
#pragma unroll
    for (int i = 0; i < 2; ++i)
#pragma unroll
      for (int j = 0; j < 2; ++j) acc[i][j] = wmma_bf16(a[i], b[j], acc[i][j]);

    wait_async0();   // next buffer landed in LDS
    __syncthreads(); // everyone done reading current buffer
  }

  const int half = lane >> 4, nlo = lane & 15;
#pragma unroll
  for (int i = 0; i < 2; ++i)
#pragma unroll
    for (int j = 0; j < 2; ++j)
#pragma unroll
      for (int r = 0; r < 8; ++r) {
        size_t row = (size_t)(m0b + m0 + 16 * i + r + 8 * half);
        size_t col = (size_t)(n0b + n0 + 16 * j + nlo);
        if (OUT_BF16)
          ((bf16*)Cout)[row * N + col] = (bf16)acc[i][j][r];
        else
          ((float*)Cout)[row * N + col] = acc[i][j][r];
      }
}

// ---------------------------------------------------------------------------
// Flash attention (causal), TRANSPOSED-score formulation.
// 128 threads = 4 waves; each wave owns 32 queries x dh=64.
//   S^T = K * Q^T   (A = K tile [kv][dh], B = Q [q][dh]; both natural layouts)
//   -> a query owns a COLUMN of S^T: its 16 scores/tile sit in TWO lanes (l, l+16),
//      so softmax needs only in-lane reductions + one shfl_xor(16).
//   O^T = V^T * P^T (A = V^T rows [dh][kv], B = P [q][kv] via LDS bounce)
//   -> per-query rescale / 1/l normalize are pure per-lane ops.
// q pre-scaled by dh^-0.5 at reshape time.
// ---------------------------------------------------------------------------
__global__ __launch_bounds__(128) void flash_attn(const bf16* __restrict__ qh,
                                                  const bf16* __restrict__ kh,
                                                  const bf16* __restrict__ vt,
                                                  bf16* __restrict__ attn_out) {
  constexpr int SEQ = 2048, DHD = 64;
  const int lane = threadIdx.x & 31;
  const int wv = threadIdx.x >> 5;
  const int half = lane >> 4, nlo = lane & 15;
  const int bh = blockIdx.y;
  const int b = bh >> 4, h = bh & 15;
  const bf16* Q = qh + (size_t)bh * SEQ * DHD;   // [token][dh]
  const bf16* Km = kh + (size_t)bh * SEQ * DHD;  // [token][dh]
  const bf16* Vt = vt + (size_t)bh * DHD * SEQ;  // [dh][token]
  const int q0 = (blockIdx.x * 4 + wv) * 32;

  __shared__ __align__(16) bf16 pst[4][32][32];  // per-wave P staging [q][kv]
  bf16(*P)[32] = pst[wv];

  // Q resident as the B operand of S^T = K Q^T: 2 q-tiles x 2 dh-chunks
  v16bf qb[2][2];
#pragma unroll
  for (int nt = 0; nt < 2; ++nt)
#pragma unroll
    for (int kc = 0; kc < 2; ++kc)
      qb[nt][kc] = load_b_frag(Q, DHD, q0 + 16 * nt, kc * 32, lane);

  v8f Ot[4][2] = {};                       // O^T accumulators [dh-tile][q-tile]
  float mq[2] = {-1e30f, -1e30f};          // running max per owned query column
  float lq[2] = {0.f, 0.f};                // running sum per owned query column

  for (int j0 = 0; j0 <= q0; j0 += 32) {
    // prefetch next K tile: each of 32 lanes pulls one 128B row
    if (j0 + 32 <= q0)
      __builtin_prefetch(Km + (size_t)(j0 + 32 + lane) * DHD, 0, 0);

    // ---- S^T = K Q^T ----
    v16bf ka[2][2];
#pragma unroll
    for (int mt = 0; mt < 2; ++mt)
#pragma unroll
      for (int kc = 0; kc < 2; ++kc)
        ka[mt][kc] = load_a_frag(Km, DHD, j0 + 16 * mt, kc * 32, lane);

    v8f St[2][2] = {};  // [kv-tile][q-tile]
#pragma unroll
    for (int mt = 0; mt < 2; ++mt)
#pragma unroll
      for (int nt = 0; nt < 2; ++nt)
#pragma unroll
        for (int kc = 0; kc < 2; ++kc)
          St[mt][nt] = wmma_bf16(ka[mt][kc], qb[nt][kc], St[mt][nt]);

    // ---- causal mask (diagonal block only; wave-uniform branch) ----
    if (j0 == q0) {
#pragma unroll
      for (int mt = 0; mt < 2; ++mt)
#pragma unroll
        for (int nt = 0; nt < 2; ++nt)
#pragma unroll
          for (int r = 0; r < 8; ++r) {
            int kv = j0 + 16 * mt + r + 8 * half;
            int qi = q0 + 16 * nt + nlo;
            if (kv > qi) St[mt][nt][r] = -1e30f;
          }
    }

    // ---- online softmax: queries are columns -> per-lane reductions ----
#pragma unroll
    for (int nt = 0; nt < 2; ++nt) {
      float t = -1e30f;
#pragma unroll
      for (int mt = 0; mt < 2; ++mt)
#pragma unroll
        for (int r = 0; r < 8; ++r) t = fmaxf(t, St[mt][nt][r]);
      t = fmaxf(t, __shfl_xor(t, 16, 32));  // combine the two kv half-tiles
      float mn = fmaxf(mq[nt], t);
      float sc = __expf(mq[nt] - mn);
      mq[nt] = mn;

      float rs = 0.f;
#pragma unroll
      for (int mt = 0; mt < 2; ++mt) {
        v8bf pk;
#pragma unroll
        for (int r = 0; r < 8; ++r) {
          float p = __expf(St[mt][nt][r] - mn);
          rs += p;
          pk[r] = (bf16)p;
        }
        // P[q][kv]: 8 contiguous kv values -> one 16B LDS store
        *(v8bf*)&P[16 * nt + nlo][16 * mt + 8 * half] = pk;
      }
      rs += __shfl_xor(rs, 16, 32);
      lq[nt] = lq[nt] * sc + rs;

#pragma unroll
      for (int dt = 0; dt < 4; ++dt)
#pragma unroll
        for (int r = 0; r < 8; ++r) Ot[dt][nt][r] *= sc;
    }

    // ---- O^T += V^T P^T  (B operand = P [q][kv] from LDS) ----
    v16bf pb[2];
    pb[0] = load_b_frag(&P[0][0], 32, 0, 0, lane);
    pb[1] = load_b_frag(&P[0][0], 32, 16, 0, lane);
#pragma unroll
    for (int dt = 0; dt < 4; ++dt) {
      v16bf va = load_a_frag(Vt, SEQ, 16 * dt, j0, lane);
      Ot[dt][0] = wmma_bf16(va, pb[0], Ot[dt][0]);
      Ot[dt][1] = wmma_bf16(va, pb[1], Ot[dt][1]);
    }
  }

  // ---- normalize (per-lane) and store: 16B vector stores, contiguous dh ----
#pragma unroll
  for (int nt = 0; nt < 2; ++nt) {
    const int qi = q0 + 16 * nt + nlo;
    const size_t token = (size_t)b * SEQ + qi;
    const float inv = 1.f / lq[nt];
#pragma unroll
    for (int dt = 0; dt < 4; ++dt) {
      v8bf ov;
#pragma unroll
      for (int r = 0; r < 8; ++r) ov[r] = (bf16)(Ot[dt][nt][r] * inv);
      *(v8bf*)&attn_out[token * 1024 + h * 64 + 16 * dt + 8 * half] = ov;
    }
  }
}

// ---------------------------------------------------------------------------
// Host-side orchestration
// ---------------------------------------------------------------------------
extern "C" void kernel_launch(void* const* d_in, const int* in_sizes, int n_in,
                              void* d_out, int out_size, void* d_ws, size_t ws_size,
                              hipStream_t stream) {
  (void)in_sizes; (void)n_in; (void)out_size; (void)ws_size;
  const float* x      = (const float*)d_in[0];  // [4,2048,1024]
  const float* w_qkv  = (const float*)d_in[1];  // [1024,3072]
  const float* w_proj = (const float*)d_in[2];  // [1024,1024]
  float* out = (float*)d_out;                   // [4,2048,1024]

  constexpr size_t B = 4, SEQ = 2048, D = 1024, H = 16;
  constexpr size_t TOK = B * SEQ;  // 8192

  // workspace layout (all bf16, ~143 MB)
  bf16* xb     = (bf16*)d_ws;                 // [8192,1024]
  bf16* wqkvT  = xb + TOK * D;                // [3072,1024]
  bf16* wprojT = wqkvT + 3 * D * D;           // [1024,1024]
  bf16* qkv    = wprojT + D * D;              // [8192,3072]
  bf16* qh     = qkv + TOK * 3 * D;           // [64,2048,64]
  bf16* kh     = qh + TOK * D;
  bf16* vtp    = kh + TOK * D;
  bf16* attn   = vtp + TOK * D;               // [8192,1024]

  {
    long long n = (long long)TOK * D;
    cast_f32_bf16<<<(unsigned)((n + 255) / 256), 256, 0, stream>>>(x, xb, n);
  }
  {
    long long n = (long long)D * 3 * D;
    transpose_f32_bf16<<<(unsigned)((n + 255) / 256), 256, 0, stream>>>(
        w_qkv, wqkvT, (int)D, (int)(3 * D));
  }
  {
    long long n = (long long)D * D;
    transpose_f32_bf16<<<(unsigned)((n + 255) / 256), 256, 0, stream>>>(
        w_proj, wprojT, (int)D, (int)D);
  }

  // QKV projection: [8192,1024] x [1024,3072] -> bf16 [8192,3072]
  tile_gemm<true><<<dim3((unsigned)(3 * D / 64), (unsigned)(TOK / 128)), 256, 0,
                    stream>>>(xb, wqkvT, qkv, (int)TOK, (int)(3 * D), (int)D);

  {
    long long n = (long long)TOK * D;
    reshape_qkv<<<(unsigned)((n + 255) / 256), 256, 0, stream>>>(qkv, qh, kh, vtp);
  }

  // Flash attention: grid (q-tiles/4, b*h), 4 waves/block
  flash_attn<<<dim3((unsigned)(SEQ / 32 / 4), (unsigned)(B * H)), 128, 0, stream>>>(
      qh, kh, vtp, attn);

  // Output projection: [8192,1024] x [1024,1024] -> fp32 out
  tile_gemm<false><<<dim3((unsigned)(D / 64), (unsigned)(TOK / 128)), 256, 0,
                     stream>>>(attn, wprojT, out, (int)TOK, (int)D, (int)D);
}
